// LinearAttention_17214228922443
// MI455X (gfx1250) — matrix operand order repaired and checked by reading, not compiled, not run on hardware
//
#include <hip/hip_runtime.h>
#include <hip/hip_bf16.h>

// ============================================================================
// MI455X (gfx1250) linear-attention block.
// GEMMs run on v_wmma_f32_16x16x32_bf16 with bf16x3 fp32 emulation; all GEMM
// operands are pre-split once into bf16 hi/lo planes (inner loops are pure
// bf16 loads + WMMA). Bucket/attention tiles are staged into LDS with the
// Tensor Data Mover (tensor_load_to_lds + s_wait_tensorcnt).
//
// Workspace layout:
//   fp32 : Q(8192x2048) K(8192x1024) V(8192x1024) S(1024x128x128) kcum(1024x128)
//   bf16 : xh/xl (8192x2048, reused as attn hi/lo planes), wq/wk/wv/wo hi/lo
// Total ~319 MB.
// ============================================================================

typedef __attribute__((ext_vector_type(16))) __bf16 v16bf;
typedef __attribute__((ext_vector_type(8)))  __bf16 v8bf;
typedef __attribute__((ext_vector_type(4)))  __bf16 v4bf;
typedef __attribute__((ext_vector_type(8)))  float  v8f;
typedef __attribute__((ext_vector_type(4)))  float  v4f;
typedef __attribute__((ext_vector_type(4)))  unsigned u32x4;
typedef __attribute__((ext_vector_type(4)))  int      i32x4;
typedef __attribute__((ext_vector_type(8)))  int      i32x8;

// ---------------- bf16 split helpers (bf16x3 fp32 emulation) ----------------
__device__ __forceinline__ unsigned short f32_to_bf16_rne(float f) {
  unsigned u = __builtin_bit_cast(unsigned, f);
  unsigned r = u + 0x7FFFu + ((u >> 16) & 1u);
  return (unsigned short)(r >> 16);
}
__device__ __forceinline__ float bf16_bits_to_f32(unsigned short h) {
  return __builtin_bit_cast(float, (unsigned)h << 16);
}
struct bfpair { __bf16 hi, lo; };
__device__ __forceinline__ bfpair split_bf16(float f) {
  unsigned short h = f32_to_bf16_rne(f);
  float rem = f - bf16_bits_to_f32(h);
  unsigned short l = f32_to_bf16_rne(rem);
  bfpair p;
  p.hi = __builtin_bit_cast(__bf16, h);
  p.lo = __builtin_bit_cast(__bf16, l);
  return p;
}

// D += A*B with bf16x3: hi*hi + hi*lo + lo*hi (lo*lo dropped, ~2^-16 rel err)
__device__ __forceinline__ v8f wmma3(v16bf ah, v16bf al, v16bf bh, v16bf bl, v8f acc) {
  acc = __builtin_amdgcn_wmma_f32_16x16x32_bf16(false, ah, false, bh, (short)0, acc, false, false);
  acc = __builtin_amdgcn_wmma_f32_16x16x32_bf16(false, ah, false, bl, (short)0, acc, false, false);
  acc = __builtin_amdgcn_wmma_f32_16x16x32_bf16(false, al, false, bh, (short)0, acc, false, false);
  return acc;
}

// ---------------- Tensor Data Mover: 2D tile -> LDS ----------------
// D# per CDNA5 ISA 8.3/8.4: count=1, type=2(image), data_size=4B,
// tile = rows x dim0 elements, row stride = stride elements.
__device__ __forceinline__ void tdm_load_2d(unsigned lds_off, const void* gptr,
                                            unsigned dim0, unsigned rows,
                                            unsigned stride) {
  unsigned long long ga = (unsigned long long)(size_t)gptr;
  u32x4 g0;
  g0[0] = 1u;                                           // count=1, user mode
  g0[1] = lds_off;                                      // lds_addr (bytes)
  g0[2] = (unsigned)ga;                                 // global_addr[31:0]
  g0[3] = (unsigned)((ga >> 32) & 0x01FFFFFFu) | (2u << 30);  // addr[56:32] | type=2
  i32x8 g1;
  g1[0] = (int)(2u << 16);                              // data_size = 4 bytes
  g1[1] = (int)((dim0 & 0xFFFFu) << 16);                // tensor_dim0[15:0]
  g1[2] = (int)((dim0 >> 16) | ((rows & 0xFFFFu) << 16));   // dim0[31:16]|dim1[15:0]
  g1[3] = (int)((rows >> 16) | ((dim0 & 0xFFFFu) << 16));   // dim1[31:16]|tile_dim0
  g1[4] = (int)(rows & 0xFFFFu);                        // tile_dim1 (tile_dim2=0)
  g1[5] = (int)stride;                                  // tensor_dim0_stride[31:0]
  g1[6] = 0;
  g1[7] = 0;
  i32x4 z4 = {0, 0, 0, 0};
#if __clang_major__ >= 23
  i32x8 z8 = {0, 0, 0, 0, 0, 0, 0, 0};
  __builtin_amdgcn_tensor_load_to_lds(g0, g1, z4, z4, z8, 0);
#else
  __builtin_amdgcn_tensor_load_to_lds(g0, g1, z4, z4, 0);
#endif
}
__device__ __forceinline__ unsigned lds_off32(const void* p) {
  return (unsigned)(size_t)p;  // LDS generic addr: low 32 bits = LDS offset
}

// ---------------- bf16-plane fragment loaders (wave32 ISA layouts) ----------
// A 16x32 from row-major bf16 [M][K]; lane&15 = row; lane half picks
// K {0..7,16..23} vs {8..15,24..31}.
__device__ __forceinline__ v16bf fragA_bf(const __bf16* src, int ld) {
  int lane = threadIdx.x & 31;
  const __bf16* p = src + (size_t)(lane & 15) * ld + ((lane >> 4) << 3);
  v8bf a0 = *(const v8bf*)(p);
  v8bf a1 = *(const v8bf*)(p + 16);
  return __builtin_shufflevector(a0, a1, 0, 1, 2, 3, 4, 5, 6, 7,
                                 8, 9, 10, 11, 12, 13, 14, 15);
}
// B 32x16 from row-major bf16 [N][K]; lane&15 = col n; lane half picks
// K {0..7,16..23} vs {8..15,24..31}.
__device__ __forceinline__ v16bf fragB_bf(const __bf16* src, int ld) {
  int lane = threadIdx.x & 31;
  const __bf16* p = src + (size_t)(lane & 15) * ld + ((lane >> 4) << 3);
  v8bf b0 = *(const v8bf*)(p);
  v8bf b1 = *(const v8bf*)(p + 16);
  return __builtin_shufflevector(b0, b1, 0, 1, 2, 3, 4, 5, 6, 7,
                                 8, 9, 10, 11, 12, 13, 14, 15);
}

// ---------------- fp32-source fragment loaders (LDS tiles, split on load) ---
__device__ __forceinline__ void fragA_rm(const float* src, int ld, v16bf& hi, v16bf& lo) {
  int lane = threadIdx.x & 31;
  const float* p = src + (size_t)(lane & 15) * ld + ((lane >> 4) << 3);
#pragma unroll
  for (int i = 0; i < 8; ++i) {
    bfpair s0 = split_bf16(p[i]);
    bfpair s1 = split_bf16(p[i + 16]);
    hi[i] = s0.hi; lo[i] = s0.lo;
    hi[i + 8] = s1.hi; lo[i + 8] = s1.lo;
  }
}
__device__ __forceinline__ void fragA_km(const float* src, int ld, v16bf& hi, v16bf& lo) {
  int lane = threadIdx.x & 31;
  int m = lane & 15, koff = (lane >> 4) << 3;
  const float* p = src + (size_t)koff * ld + m;
#pragma unroll
  for (int i = 0; i < 8; ++i) {
    bfpair s0 = split_bf16(p[(size_t)i * ld]);
    bfpair s1 = split_bf16(p[(size_t)(i + 16) * ld]);
    hi[i] = s0.hi; lo[i] = s0.lo;
    hi[i + 8] = s1.hi; lo[i + 8] = s1.lo;
  }
}
__device__ __forceinline__ void fragB_kn(const float* src, int ld, v16bf& hi, v16bf& lo) {
  int lane = threadIdx.x & 31;
  const float* p = src + (size_t)((lane >> 4) << 3) * ld + (lane & 15);
#pragma unroll
  for (int i = 0; i < 16; ++i) { bfpair s = split_bf16(p[(size_t)(i + (i >> 3) * 8) * ld]); hi[i] = s.hi; lo[i] = s.lo; }
}
// C/D 16x16 f32 store: VGPR r -> M = r (+8 for upper lane half), N = lane&15.
__device__ __forceinline__ void store_tile(float* dst, int ld, v8f acc) {
  int lane = threadIdx.x & 31;
  int n = lane & 15, m0 = (lane >> 4) << 3;
#pragma unroll
  for (int r = 0; r < 8; ++r) dst[(size_t)(m0 + r) * ld + n] = acc[r];
}

// ============================================================================
// Pre-split fp32 -> bf16 hi/lo planes (one pass; GEMMs then load pure bf16).
// ============================================================================
__global__ __launch_bounds__(256) void split_planes(const float* __restrict__ src,
                                                    __bf16* __restrict__ hi,
                                                    __bf16* __restrict__ lo, int n) {
  int i = (blockIdx.x * 256 + threadIdx.x) * 4;
  if (i >= n) return;
  v4f v = *(const v4f*)(src + i);
  v4bf h, l;
#pragma unroll
  for (int j = 0; j < 4; ++j) { bfpair s = split_bf16(v[j]); h[j] = s.hi; l[j] = s.lo; }
  *(v4bf*)(hi + i) = h;
  *(v4bf*)(lo + i) = l;
}

// ============================================================================
// GEMM: C[M,N] = A[M,K] * B[N,K]^T from bf16 hi/lo planes.
// grid = (N/256, M/128), 256 threads = 8 waves (2 along M x 4 along N),
// wave tile 64x64: 16 accumulators, 48 WMMA per 32-deep k-step.
// Operands stream from the 192 MB L2 (x planes 67 MB + weights fit).
// ============================================================================
__global__ __launch_bounds__(256) void gemm_bf3(const __bf16* __restrict__ Ah,
                                                const __bf16* __restrict__ Al,
                                                const __bf16* __restrict__ Bh,
                                                const __bf16* __restrict__ Bl,
                                                float* __restrict__ C, int N, int K) {
  int wave = threadIdx.x >> 5;
  int wm = wave >> 2, wn = wave & 3;
  size_t m0 = (size_t)blockIdx.y * 128 + (size_t)wm * 64;
  size_t n0 = (size_t)blockIdx.x * 256 + (size_t)wn * 64;
  v8f zero = {};
  v8f acc[4][4];
#pragma unroll
  for (int mt = 0; mt < 4; ++mt)
#pragma unroll
    for (int nt = 0; nt < 4; ++nt) acc[mt][nt] = zero;

  for (int k = 0; k < K; k += 32) {
    v16bf ah[4], al[4], bh[4], bl[4];
#pragma unroll
    for (int mt = 0; mt < 4; ++mt) {
      size_t off = (m0 + mt * 16) * K + k;
      ah[mt] = fragA_bf(Ah + off, K);
      al[mt] = fragA_bf(Al + off, K);
    }
#pragma unroll
    for (int nt = 0; nt < 4; ++nt) {
      size_t off = (n0 + nt * 16) * K + k;
      bh[nt] = fragB_bf(Bh + off, K);
      bl[nt] = fragB_bf(Bl + off, K);
    }
#pragma unroll
    for (int mt = 0; mt < 4; ++mt)
#pragma unroll
      for (int nt = 0; nt < 4; ++nt)
        acc[mt][nt] = wmma3(ah[mt], al[mt], bh[nt], bl[nt], acc[mt][nt]);
  }
#pragma unroll
  for (int mt = 0; mt < 4; ++mt)
#pragma unroll
    for (int nt = 0; nt < 4; ++nt)
      store_tile(C + (m0 + mt * 16) * N + n0 + nt * 16, N, acc[mt][nt]);
}

// ============================================================================
// RoPE + activation. One wave per (token, head) row of 128 values.
// do_softmax=1: q = softmax(rot(q)) * 128^-0.5 ; else k = exp(rot(k)).
// ============================================================================
__global__ __launch_bounds__(256) void rotary_act(float* __restrict__ buf,
                                                  const float* __restrict__ freqs,
                                                  int hcount, int do_softmax) {
  int lane = threadIdx.x & 31;
  int row = blockIdx.x * 8 + (threadIdx.x >> 5);
  int t = row / hcount;
  int h = row - t * hcount;
  int pos = t & 4095;
  float* base = buf + (size_t)t * (hcount * 128) + h * 128;
  v4f v = *(const v4f*)(base + 4 * lane);
  v4f cs = *(const v4f*)(freqs + (size_t)pos * 128 + 4 * lane);  // c0,s0,c1,s1
  float o0 = v[0] * cs[0] - v[1] * cs[1];
  float o1 = v[0] * cs[1] + v[1] * cs[0];
  float o2 = v[2] * cs[2] - v[3] * cs[3];
  float o3 = v[2] * cs[3] + v[3] * cs[2];
  if (do_softmax) {
    float m = fmaxf(fmaxf(o0, o1), fmaxf(o2, o3));
#pragma unroll
    for (int off = 16; off >= 1; off >>= 1) m = fmaxf(m, __shfl_xor(m, off, 32));
    float e0 = expf(o0 - m), e1 = expf(o1 - m), e2 = expf(o2 - m), e3 = expf(o3 - m);
    float s = e0 + e1 + e2 + e3;
#pragma unroll
    for (int off = 16; off >= 1; off >>= 1) s += __shfl_xor(s, off, 32);
    float sc = 0.08838834764831845f / s;  // 128^-0.5 / sum
    o0 = e0 * sc; o1 = e1 * sc; o2 = e2 * sc; o3 = e3 * sc;
  } else {
    o0 = expf(o0); o1 = expf(o1); o2 = expf(o2); o3 = expf(o3);
  }
  v4f o; o[0] = o0; o[1] = o1; o[2] = o2; o[3] = o3;
  *(v4f*)(base + 4 * lane) = o;
}

// ============================================================================
// Per-bucket context: S[b,kvh,u][d][e] = sum_{n in bucket} k[n,d] * v[n,e].
// One block per (b,kvh,u); k/v tiles TDM-loaded into LDS; WMMA 128x128 out.
// ============================================================================
__global__ __launch_bounds__(256) void bucket_ctx(const float* __restrict__ Kb,
                                                  const float* __restrict__ Vb,
                                                  float* __restrict__ S) {
  __shared__ float lk[64 * 128];
  __shared__ float lv[64 * 128];
  int u = blockIdx.x & 63;
  int kvh = (blockIdx.x >> 6) & 7;
  int b = blockIdx.x >> 9;
  size_t tbase = (size_t)b * 4096 + (size_t)u * 64;
  if ((threadIdx.x >> 5) == 0) {  // wave 0 drives the Tensor Data Mover
    tdm_load_2d(lds_off32(lk), Kb + tbase * 1024 + kvh * 128, 128, 64, 1024);
    tdm_load_2d(lds_off32(lv), Vb + tbase * 1024 + kvh * 128, 128, 64, 1024);
    __builtin_amdgcn_s_wait_tensorcnt(0);
  }
  __syncthreads();

  int wave = threadIdx.x >> 5;
  int d0 = wave * 16;
  v8f zero = {};
  v8f acc[8];
#pragma unroll
  for (int et = 0; et < 8; ++et) acc[et] = zero;
#pragma unroll
  for (int ks = 0; ks < 64; ks += 32) {
    v16bf ah, al;
    fragA_km(lk + ks * 128 + d0, 128, ah, al);  // A[d,n] = k[n,d]
#pragma unroll
    for (int et = 0; et < 8; ++et) {
      v16bf bh, bl;
      fragB_kn(lv + ks * 128 + et * 16, 128, bh, bl);  // B[n,e] = v[n,e]
      acc[et] = wmma3(ah, al, bh, bl, acc[et]);
    }
  }
  float* Sout = S + ((size_t)(b * 8 + kvh) * 64 + u) * 16384;
#pragma unroll
  for (int et = 0; et < 8; ++et)
    store_tile(Sout + (size_t)d0 * 128 + et * 16, 128, acc[et]);
}

// Exclusive prefix over buckets of S, in place. One thread per (b,kvh,d,e).
__global__ __launch_bounds__(256) void ctx_cumsum(float* __restrict__ S) {
  int idx = blockIdx.x * 256 + threadIdx.x;  // 262144 threads
  int bh = idx >> 14;
  int de = idx & 16383;
  float* p = S + (size_t)bh * 64 * 16384 + de;
  float run = 0.f;
  for (int u = 0; u < 64; ++u) {
    float t = p[(size_t)u * 16384];
    p[(size_t)u * 16384] = run;
    run += t;
  }
}

// kcum[b,kvh,u][d] = exclusive prefix of bucket k-sums. One thread/(b,kvh,d).
__global__ __launch_bounds__(256) void k_cumsum(const float* __restrict__ Kb,
                                                float* __restrict__ kcum) {
  int idx = blockIdx.x * 256 + threadIdx.x;  // 2048 threads
  int b = idx >> 10, kvh = (idx >> 7) & 7, d = idx & 127;
  float run = 0.f;
  for (int u = 0; u < 64; ++u) {
    kcum[((size_t)(b * 8 + kvh) * 64 + u) * 128 + d] = run;
    size_t t0 = (size_t)b * 4096 + (size_t)u * 64;
    for (int n = 0; n < 64; ++n) run += Kb[(t0 + n) * 1024 + kvh * 128 + d];
  }
}

// ============================================================================
// Attention apply: out[n,e] = (q[n,:] @ S_u) * D_inv[n],
// D[n] = max(dot(kcum_u, q[n,:]), 1e-6). One block per (b,h,u); q and S_u
// TDM-loaded into LDS; result written directly as bf16 hi/lo planes for the
// final projection GEMM.
// ============================================================================
__global__ __launch_bounds__(256) void attn_apply(const float* __restrict__ Q,
                                                  const float* __restrict__ S,
                                                  const float* __restrict__ kcum,
                                                  __bf16* __restrict__ AH,
                                                  __bf16* __restrict__ AL) {
  __shared__ float qs[64 * 128];
  __shared__ float ss[128 * 128];
  __shared__ float dinv[64];
  int u = blockIdx.x & 63;
  int h = (blockIdx.x >> 6) & 15;
  int b = blockIdx.x >> 10;
  int kvh = h >> 1;
  size_t tbase = (size_t)b * 4096 + (size_t)u * 64;
  const float* Sbase = S + ((size_t)(b * 8 + kvh) * 64 + u) * 16384;
  if ((threadIdx.x >> 5) == 0) {
    tdm_load_2d(lds_off32(qs), Q + tbase * 2048 + h * 128, 128, 64, 2048);
    tdm_load_2d(lds_off32(ss), Sbase, 16384, 1, 16384);
    __builtin_amdgcn_s_wait_tensorcnt(0);
  }
  __syncthreads();

  int wave = threadIdx.x >> 5, lane = threadIdx.x & 31;
  const float* kc = kcum + ((size_t)(b * 8 + kvh) * 64 + u) * 128;
  for (int j = 0; j < 8; ++j) {
    int rrow = wave * 8 + j;
    float p = 0.f;
#pragma unroll
    for (int c = 0; c < 4; ++c) {
      int d = lane * 4 + c;
      p += kc[d] * qs[rrow * 128 + d];
    }
#pragma unroll
    for (int off = 16; off >= 1; off >>= 1) p += __shfl_xor(p, off, 32);
    if (lane == 0) dinv[rrow] = 1.0f / fmaxf(p, 1e-6f);
  }
  __syncthreads();

  int wm = wave >> 1, wn = wave & 1;
  int m0 = wm * 16, n0 = wn * 64;
  v8f zero = {};
  v8f acc[4];
#pragma unroll
  for (int nt = 0; nt < 4; ++nt) acc[nt] = zero;
#pragma unroll
  for (int ks = 0; ks < 128; ks += 32) {
    v16bf ah, al;
    fragA_rm(qs + m0 * 128 + ks, 128, ah, al);
#pragma unroll
    for (int nt = 0; nt < 4; ++nt) {
      v16bf bh, bl;
      fragB_kn(ss + ks * 128 + n0 + nt * 16, 128, bh, bl);
      acc[nt] = wmma3(ah, al, bh, bl, acc[nt]);
    }
  }
  int mrow0 = m0 + ((lane >> 4) << 3);
  int col = lane & 15;
#pragma unroll
  for (int nt = 0; nt < 4; ++nt)
#pragma unroll
    for (int r = 0; r < 8; ++r) {
      int m = mrow0 + r;
      float v = acc[nt][r] * dinv[m];
      bfpair s = split_bf16(v);
      size_t o = (tbase + m) * 2048 + h * 128 + n0 + nt * 16 + col;
      AH[o] = s.hi;
      AL[o] = s.lo;
    }
}

// ============================================================================
extern "C" void kernel_launch(void* const* d_in, const int* in_sizes, int n_in,
                              void* d_out, int out_size, void* d_ws, size_t ws_size,
                              hipStream_t stream) {
  const float* x     = (const float*)d_in[0];  // (2,4096,2048)
  const float* freqs = (const float*)d_in[1];  // (4096,64,2)
  const float* wq    = (const float*)d_in[2];  // (2048,2048)
  const float* wk    = (const float*)d_in[3];  // (1024,2048)
  const float* wv    = (const float*)d_in[4];  // (1024,2048)
  const float* wo    = (const float*)d_in[5];  // (2048,2048)
  float* out = (float*)d_out;                  // (2,4096,2048)

  // fp32 region
  float* Q    = (float*)d_ws;            // 16777216 floats
  float* Kb   = Q + 16777216;            //  8388608
  float* Vb   = Kb + 8388608;            //  8388608
  float* S    = Vb + 8388608;            // 16777216
  float* kcum = S + 16777216;            //   131072
  // bf16 plane region
  __bf16* bfbase = (__bf16*)(kcum + 131072);
  __bf16* xh  = bfbase;                  // 16777216 (reused as attn hi plane)
  __bf16* xl  = xh + 16777216;           // 16777216 (reused as attn lo plane)
  __bf16* wqh = xl + 16777216;           //  4194304
  __bf16* wql = wqh + 4194304;
  __bf16* wkh = wql + 4194304;           //  2097152
  __bf16* wkl = wkh + 2097152;
  __bf16* wvh = wkl + 2097152;           //  2097152
  __bf16* wvl = wvh + 2097152;
  __bf16* woh = wvl + 2097152;           //  4194304
  __bf16* wol = woh + 4194304;

  dim3 blk(256);
  // pre-split all GEMM operands into bf16 hi/lo planes
  split_planes<<<16384, blk, 0, stream>>>(x, xh, xl, 16777216);
  split_planes<<<4096, blk, 0, stream>>>(wq, wqh, wql, 4194304);
  split_planes<<<2048, blk, 0, stream>>>(wk, wkh, wkl, 2097152);
  split_planes<<<2048, blk, 0, stream>>>(wv, wvh, wvl, 2097152);
  split_planes<<<4096, blk, 0, stream>>>(wo, woh, wol, 4194304);
  // QKV projections (x @ W^T)
  gemm_bf3<<<dim3(8, 64), blk, 0, stream>>>(xh, xl, wqh, wql, Q, 2048, 2048);
  gemm_bf3<<<dim3(4, 64), blk, 0, stream>>>(xh, xl, wkh, wkl, Kb, 1024, 2048);
  gemm_bf3<<<dim3(4, 64), blk, 0, stream>>>(xh, xl, wvh, wvl, Vb, 1024, 2048);
  // RoPE + softmax (Q) / exp (K)
  rotary_act<<<16384, blk, 0, stream>>>(Q, freqs, 16, 1);
  rotary_act<<<8192, blk, 0, stream>>>(Kb, freqs, 8, 0);
  // bucketed context + exclusive prefixes
  bucket_ctx<<<1024, blk, 0, stream>>>(Kb, Vb, S);
  ctx_cumsum<<<1024, blk, 0, stream>>>(S);
  k_cumsum<<<8, blk, 0, stream>>>(Kb, kcum);
  // attention apply -> bf16 hi/lo planes (x planes are dead now; reuse them)
  attn_apply<<<2048, blk, 0, stream>>>(Q, S, kcum, xh, xl);
  // output projection
  gemm_bf3<<<dim3(8, 64), blk, 0, stream>>>(xh, xl, woh, wol, out, 2048, 2048);
}
